// VisionSdpaAttention_48069273977253
// MI455X (gfx1250) — matrix operand order repaired and checked by reading, not compiled, not run on hardware
//
#include <hip/hip_runtime.h>
#include <math.h>

typedef __attribute__((ext_vector_type(16))) __bf16 v16bf;
typedef __attribute__((ext_vector_type(8)))  __bf16 v8bf;
typedef __attribute__((ext_vector_type(8)))  float  v8f;
typedef __attribute__((ext_vector_type(4)))  int    v4i;

#define H 16
#define D 128
#define NEG_INF (-__builtin_inff())

#define KF_PITCH 132   // f32 pitch: row stride 528B (16B aligned), bank spread 4*col
#define VT_PITCH 40    // bf16 pitch: row stride 80B, bank spread (20*col)%64 (disjoint)

#if defined(__has_builtin)
# if __has_builtin(__builtin_amdgcn_global_load_async_to_lds_b128)
#  define HAVE_ASYNC_LDS 1
# else
#  define HAVE_ASYNC_LDS 0
# endif
# if __has_builtin(__builtin_amdgcn_s_wait_asynccnt)
#  define HAVE_WAIT_ASYNC 1
# else
#  define HAVE_WAIT_ASYNC 0
# endif
#else
# define HAVE_ASYNC_LDS 0
# define HAVE_WAIT_ASYNC 0
#endif

#if HAVE_ASYNC_LDS
typedef __attribute__((address_space(1))) v4i* gv4i_p;   // global int4*
typedef __attribute__((address_space(3))) v4i* lv4i_p;   // LDS int4*
#endif

__device__ __forceinline__ void async_wait0() {
#if HAVE_ASYNC_LDS
# if HAVE_WAIT_ASYNC
    __builtin_amdgcn_s_wait_asynccnt(0);
# else
    asm volatile("s_wait_asynccnt 0" ::: "memory");
# endif
#endif
}

__launch_bounds__(128, 1)
__global__ void vis_sdpa_fa_kernel(const float* __restrict__ q,
                                   const float* __restrict__ k,
                                   const float* __restrict__ v,
                                   const int*   __restrict__ cu, int ncu,
                                   float* __restrict__ out,
                                   int total, float scale) {
    __shared__ __align__(16) float  Kf[2][32][KF_PITCH];  // K chunk, f32 (async dest), dbl-buf
    __shared__ __align__(16) __bf16 Vt[2][D][VT_PITCH];   // V chunk, bf16 transposed, dbl-buf
    __shared__ __align__(16) __bf16 Pt[4][16][32];        // per-wave P staging

    const int tid    = threadIdx.x;
    const int lane   = tid & 31;
    const int wid    = tid >> 5;
    const int head   = blockIdx.y;
    const int qb0    = blockIdx.x * 64;
    const int qt     = qb0 + wid * 16;

    const int col    = lane & 15;
    const int hi16   = lane >> 4;
    const int rowoff = hi16 * 8;

    // ---- per-row segment bounds [rs,re) ----
    int rs[8], re[8];
    #pragma unroll
    for (int r = 0; r < 8; ++r) {
        int g = qt + r + rowoff; if (g >= total) g = total - 1;
        int s0 = 0, e0 = total;
        for (int i = 0; i + 1 < ncu; ++i) {
            int a = cu[i], b = cu[i + 1];
            if (g >= a && g < b) { s0 = a; e0 = b; }
        }
        rs[r] = s0; re[r] = e0;
    }
    // ---- block-uniform key range (barrier-safe) ----
    int lo = total, hi = 0;
    for (int m = 0; m < 64; ++m) {
        int g = qb0 + m; if (g >= total) g = total - 1;
        for (int i = 0; i + 1 < ncu; ++i) {
            int a = cu[i], b = cu[i + 1];
            if (g >= a && g < b) { lo = min(lo, a); hi = max(hi, b); }
        }
    }

    // ---- staging helpers ----
    // K chunk: 32 rows x 512B; wave w copies rows 8w..8w+7, one instruction per row.
    auto stage_K = [&](int kb, int buf) {
        #pragma unroll
        for (int i = 0; i < 8; ++i) {
            int row = wid * 8 + i;
            int key = kb + row; if (key >= total) key = total - 1;
            const float* src = k + (size_t)key * (H * D) + head * D + lane * 4;
#if HAVE_ASYNC_LDS
            __builtin_amdgcn_global_load_async_to_lds_b128(
                (gv4i_p)src, (lv4i_p)&Kf[buf][row][lane * 4], 0, 0);
#else
            *(float4*)&Kf[buf][row][lane * 4] = *(const float4*)src;
#endif
        }
    };
    // V chunk transposed -> Vt[buf][d][key] (bf16); register transpose, all 128 threads.
    auto stage_V = [&](int kb, int buf) {
        #pragma unroll
        for (int it = 0; it < 8; ++it) {
            int u   = tid + 128 * it;
            int key = u >> 5;
            int d4  = (u & 31) * 4;
            int kg  = kb + key; if (kg >= total) kg = total - 1;
            float4 x = *(const float4*)(v + (size_t)kg * (H * D) + head * D + d4);
            Vt[buf][d4 + 0][key] = (__bf16)x.x;
            Vt[buf][d4 + 1][key] = (__bf16)x.y;
            Vt[buf][d4 + 2][key] = (__bf16)x.z;
            Vt[buf][d4 + 3][key] = (__bf16)x.w;
        }
    };

    // ---- Q tile -> 4 A-fragments (16x32 bf16), scale folded, float4 loads ----
    v16bf aQ[4];
    {
        int qr = qt + col; if (qr >= total) qr = total - 1;
        const float* qrow = q + (size_t)qr * (H * D) + head * D;
        #pragma unroll
        for (int j = 0; j < 4; ++j) {
            const int base = 32 * j + hi16 * 16;
            float4 x0 = *(const float4*)(qrow + base);
            float4 x1 = *(const float4*)(qrow + base + 4);
            float4 x2 = *(const float4*)(qrow + base + 8);
            float4 x3 = *(const float4*)(qrow + base + 12);
            aQ[j][0]  = (__bf16)(x0.x * scale); aQ[j][1]  = (__bf16)(x0.y * scale);
            aQ[j][2]  = (__bf16)(x0.z * scale); aQ[j][3]  = (__bf16)(x0.w * scale);
            aQ[j][4]  = (__bf16)(x1.x * scale); aQ[j][5]  = (__bf16)(x1.y * scale);
            aQ[j][6]  = (__bf16)(x1.z * scale); aQ[j][7]  = (__bf16)(x1.w * scale);
            aQ[j][8]  = (__bf16)(x2.x * scale); aQ[j][9]  = (__bf16)(x2.y * scale);
            aQ[j][10] = (__bf16)(x2.z * scale); aQ[j][11] = (__bf16)(x2.w * scale);
            aQ[j][12] = (__bf16)(x3.x * scale); aQ[j][13] = (__bf16)(x3.y * scale);
            aQ[j][14] = (__bf16)(x3.z * scale); aQ[j][15] = (__bf16)(x3.w * scale);
        }
    }

    const v8f zf = {0.f, 0.f, 0.f, 0.f, 0.f, 0.f, 0.f, 0.f};
    v8f acc[8];
    float rowmax[8], rowsum[8];
    #pragma unroll
    for (int nd = 0; nd < 8; ++nd) acc[nd] = zf;
    #pragma unroll
    for (int r = 0; r < 8; ++r) { rowmax[r] = NEG_INF; rowsum[r] = 0.f; }

    // ---- prologue: stage first chunk ----
    if (lo < hi) {
        stage_K(lo, 0);
        stage_V(lo, 0);
        async_wait0();
        __syncthreads();
    }

    // ================= main loop: double-buffered, 32 keys / iteration =================
    int ib = 0;
    for (int kb = lo; kb < hi; kb += 32, ib ^= 1) {
        const int nxt = kb + 32;
        if (nxt < hi) stage_K(nxt, ib ^ 1);   // async: no VGPR cost, overlaps compute

        // ---- S = Q*K^T from Kf[ib] (f32 -> bf16 at fragment build) ----
        v8f s0 = zf, s1 = zf;
        #pragma unroll
        for (int j = 0; j < 4; ++j) {
            const float* f0 = &Kf[ib][col][32 * j + hi16 * 16];
            const float* f1 = &Kf[ib][16 + col][32 * j + hi16 * 16];
            v16bf b0, b1;
            #pragma unroll
            for (int h4 = 0; h4 < 4; ++h4) {
                float4 a0 = *(const float4*)(f0 + 4 * h4);
                float4 a1 = *(const float4*)(f1 + 4 * h4);
                b0[4 * h4 + 0] = (__bf16)a0.x; b0[4 * h4 + 1] = (__bf16)a0.y;
                b0[4 * h4 + 2] = (__bf16)a0.z; b0[4 * h4 + 3] = (__bf16)a0.w;
                b1[4 * h4 + 0] = (__bf16)a1.x; b1[4 * h4 + 1] = (__bf16)a1.y;
                b1[4 * h4 + 2] = (__bf16)a1.z; b1[4 * h4 + 3] = (__bf16)a1.w;
            }
            s0 = __builtin_amdgcn_wmma_f32_16x16x32_bf16(false, aQ[j], false, b0,
                                                         (short)0, s0, false, false);
            s1 = __builtin_amdgcn_wmma_f32_16x16x32_bf16(false, aQ[j], false, b1,
                                                         (short)0, s1, false, false);
        }

        // ---- block-diagonal mask + online softmax ----
        const int key0 = kb + col, key1 = kb + 16 + col;
        float corr[8], p0v[8], p1v[8];
        #pragma unroll
        for (int r = 0; r < 8; ++r) {
            float a = (key0 >= rs[r] && key0 < re[r]) ? s0[r] : NEG_INF;
            float b = (key1 >= rs[r] && key1 < re[r]) ? s1[r] : NEG_INF;
            float cm = fmaxf(a, b);
            for (int off = 1; off < 16; off <<= 1)
                cm = fmaxf(cm, __shfl_xor(cm, off, 32));
            float nm = fmaxf(rowmax[r], cm);
            float c, e0, e1;
            if (nm == NEG_INF) { c = 1.f; e0 = 0.f; e1 = 0.f; }
            else {
                c  = __expf(rowmax[r] - nm);
                e0 = (a == NEG_INF) ? 0.f : __expf(a - nm);
                e1 = (b == NEG_INF) ? 0.f : __expf(b - nm);
            }
            float cs = e0 + e1;
            for (int off = 1; off < 16; off <<= 1)
                cs += __shfl_xor(cs, off, 32);
            rowsum[r] = rowsum[r] * c + cs;
            rowmax[r] = nm;
            corr[r] = c; p0v[r] = e0; p1v[r] = e1;
        }

        // ---- stage next V chunk into back buffer (overlaps with PV below) ----
        if (nxt < hi) stage_V(nxt, ib ^ 1);

        // ---- transpose P (C-layout) -> A-layout via per-wave LDS tile ----
        #pragma unroll
        for (int r = 0; r < 8; ++r) {
            int m = r + rowoff;
            Pt[wid][m][col]      = (__bf16)p0v[r];
            Pt[wid][m][col + 16] = (__bf16)p1v[r];
        }
        v16bf aP;
        #pragma unroll
        for (int e = 0; e < 16; ++e) {
            int kk = e + hi16 * 16;
            aP[e] = Pt[wid][col][kk];
        }

        // ---- O = O*corr + P*V from Vt[ib] (conflict-free ds_load_b128) ----
        #pragma unroll
        for (int nd = 0; nd < 8; ++nd) {
            #pragma unroll
            for (int r = 0; r < 8; ++r) acc[nd][r] *= corr[r];
            const __bf16* vsrc = &Vt[ib][16 * nd + col][hi16 * 16];
            union { v16bf w; v8bf h[2]; } bv;
            bv.h[0] = *(const v8bf*)vsrc;
            bv.h[1] = *(const v8bf*)(vsrc + 8);
            acc[nd] = __builtin_amdgcn_wmma_f32_16x16x32_bf16(false, aP, false, bv.w,
                                                              (short)0, acc[nd], false, false);
        }

        async_wait0();       // this wave's async copies for the back buffer done
        __syncthreads();     // everyone's staging visible before buffer swap
    }

    // ---- epilogue: normalize and store ----
    #pragma unroll
    for (int r = 0; r < 8; ++r) {
        int g = qt + r + rowoff;
        if (g >= total) continue;
        float inv = (rowsum[r] > 0.f) ? (1.f / rowsum[r]) : 0.f;
        float* op = out + (size_t)g * (H * D) + head * D + col;
        #pragma unroll
        for (int nd = 0; nd < 8; ++nd)
            op[16 * nd] = acc[nd][r] * inv;
    }
}

extern "C" void kernel_launch(void* const* d_in, const int* in_sizes, int n_in,
                              void* d_out, int out_size, void* d_ws, size_t ws_size,
                              hipStream_t stream) {
    const float* q  = (const float*)d_in[0];
    const float* k  = (const float*)d_in[1];
    const float* v  = (const float*)d_in[2];
    const int*   cu = (const int*)d_in[4];   // d_in[3] is bsz (== 1)
    int ncu   = in_sizes[4];
    int total = in_sizes[0] / (H * D);
    float scale = 1.0f / sqrtf((float)D);

    dim3 grid((total + 63) / 64, H);
    dim3 block(128);
    vis_sdpa_fa_kernel<<<grid, block, 0, stream>>>(q, k, v, cu, ncu,
                                                   (float*)d_out, total, scale);
}
